// SubgraphSteadyStateOperator_11278584119627
// MI455X (gfx1250) — matrix operand order repaired and checked by reading, not compiled, not run on hardware
//
#include <hip/hip_runtime.h>
#include <hip/hip_bf16.h>
#include <stdint.h>

// ---------------- types ----------------
typedef __attribute__((ext_vector_type(16))) __bf16         v16bf;
typedef __attribute__((ext_vector_type(8)))  float          v8f;
typedef __attribute__((ext_vector_type(16))) unsigned short v16u;

static __device__ inline v16bf as_bf16(v16u v) {
    union { v16u u; v16bf b; } c; c.u = v; return c.b;
}

// fp32 -> bf16 (round to nearest even), raw 16 bits
static __device__ inline unsigned short f32_to_bf16_rne(float f) {
    unsigned u = __float_as_uint(f);
    u += 0x7FFFu + ((u >> 16) & 1u);
    return (unsigned short)(u >> 16);
}
static __device__ inline float bf16_bits_to_f32(unsigned short s) {
    return __uint_as_float(((unsigned)s) << 16);
}

// D = A*B + C, bf16 in, f32 accumulate (V_WMMA_F32_16X16X32_BF16)
static __device__ inline v8f wmma_bf16(v16u a, v16u b, v8f c) {
    return __builtin_amdgcn_wmma_f32_16x16x32_bf16(
        false, as_bf16(a), false, as_bf16(b), (short)0, c, false, false);
}

// ---------------- fragment loaders ----------------
// A (16x32, 16-bit): lane<16 -> M=lane, K {k0..k0+7, k0+16..k0+23};
//                    lane>=16 -> M=lane-16, K {k0+8.., k0+24..}
static __device__ inline v16u load_fragA_bf16(const unsigned short* __restrict__ src, int ld,
                                              int m0, int k0, int lane) {
    int r = m0 + (lane & 15);
    int g = k0 + ((lane & 16) ? 8 : 0);
    const uint4* q0 = (const uint4*)(src + (size_t)r * ld + g);
    const uint4* q1 = (const uint4*)(src + (size_t)r * ld + g + 16);
    union { struct { uint4 a, b; } q; v16u v; } u;
    u.q.a = q0[0]; u.q.b = q1[0];
    return u.v;
}

// B (32x16, 16-bit): lane<16 -> N=lane, K {k0..k0+7, k0+16..k0+23}; lane>=16 -> N=lane-16, K {k0+8.., k0+24..}
static __device__ inline v16u load_fragB_bf16(const unsigned short* __restrict__ src, int ld,
                                              int n0, int k0, int lane) {
    int r = n0 + (lane & 15);
    int g = k0 + ((lane & 16) ? 8 : 0);
    const uint4* q = (const uint4*)(src + (size_t)r * ld + g);
    union { struct { uint4 a, b; } q; v16u v; } u;
    u.q.a = q[0]; u.q.b = q[2];
    return u.v;
}

// ---------------- constants ----------------
#define DX  128
#define NH  128
#define NI  384

// ---------------- small prep kernels ----------------
__global__ void __launch_bounds__(256)
prep_w_kernel(const float* __restrict__ W1, const float* __restrict__ W2,
              unsigned short* __restrict__ w1h, unsigned short* __restrict__ w1l,
              unsigned short* __restrict__ w2h, unsigned short* __restrict__ w2l) {
    int i = blockIdx.x * 256 + threadIdx.x;
    const int n1 = NH * NI, n2 = NH * NH;
    if (i < n1) {
        float f = W1[i];
        unsigned short h = f32_to_bf16_rne(f);
        w1h[i] = h;
        w1l[i] = f32_to_bf16_rne(f - bf16_bits_to_f32(h));
    } else if (i < n1 + n2) {
        int j = i - n1;
        float f = W2[j];
        unsigned short h = f32_to_bf16_rne(f);
        w2h[j] = h;
        w2l[j] = f32_to_bf16_rne(f - bf16_bits_to_f32(h));
    }
}

__global__ void __launch_bounds__(256)
zero_u32_kernel(unsigned* __restrict__ ptr, int n) {
    int i = blockIdx.x * 256 + threadIdx.x;
    if (i < n) ptr[i] = 0u;
}

// ---------------- CSR build (dst-sorted edge list) ----------------
__global__ void __launch_bounds__(256)
hist_kernel(const int* __restrict__ dst, int* __restrict__ cnt, int n_edges) {
    int i = blockIdx.x * 256 + threadIdx.x;
    if (i < n_edges) atomicAdd(&cnt[dst[i]], 1);
}

// Single-block exclusive scan of cnt[0..n) -> off[0..n], plus cursor copy.
__global__ void __launch_bounds__(1024)
scan_kernel(const int* __restrict__ cnt, int* __restrict__ off,
            int* __restrict__ cursor, int n) {
    __shared__ int sh[1024];
    int t = threadIdx.x;
    int chunk = (n + 1023) / 1024;
    int lo = t * chunk;
    int hi = lo + chunk; if (hi > n) hi = n; if (lo > n) lo = n;
    int s = 0;
    for (int i = lo; i < hi; ++i) s += cnt[i];
    sh[t] = s;
    __syncthreads();
    for (int d = 1; d < 1024; d <<= 1) {        // inclusive Hillis-Steele
        int v = (t >= d) ? sh[t - d] : 0;
        __syncthreads();
        sh[t] += v;
        __syncthreads();
    }
    int run = (t == 0) ? 0 : sh[t - 1];
    for (int i = lo; i < hi; ++i) {
        off[i] = run; cursor[i] = run;
        run += cnt[i];
    }
    if (t == 1023) off[n] = sh[1023];
}

__global__ void __launch_bounds__(256)
fill_kernel(const int* __restrict__ src, const int* __restrict__ dst,
            int* __restrict__ cursor, int* __restrict__ csr, int n_edges) {
    int i = blockIdx.x * 256 + threadIdx.x;
    if (i < n_edges) {
        int pos = atomicAdd(&cursor[dst[i]], 1);   // int atomic, cheap
        csr[pos] = src[i];
    }
}

// ---------------- Phase A: p = [x|h] @ W1[:,128:384].T  (bf16x3 WMMA) ----------------
// Block = 16 nodes, 128 threads (4 waves). A-tile converted once into LDS.
__global__ void __launch_bounds__(128)
phaseA_kernel(const float* __restrict__ x, const float* __restrict__ h,
              const unsigned short* __restrict__ w1h, const unsigned short* __restrict__ w1l,
              float* __restrict__ p) {
    __shared__ unsigned short ah_s[16][2 * DX];
    __shared__ unsigned short al_s[16][2 * DX];

    int row0 = blockIdx.x * 16;
    int tid  = threadIdx.x;
    int wave = tid >> 5;
    int lane = tid & 31;
    int n0   = wave * 32;

    // stage & split the 16x256 fp32 A-tile into LDS (once per block)
    {
        int r  = tid >> 3;                 // 0..15
        int c0 = (tid & 7) * 32;           // 0..224
        const float* sp = (c0 < DX) ? (x + (size_t)(row0 + r) * DX + c0)
                                    : (h + (size_t)(row0 + r) * DX + (c0 - DX));
#pragma unroll
        for (int j = 0; j < 32; j += 4) {
            float4 f = *(const float4*)(sp + j);
            float fa[4] = { f.x, f.y, f.z, f.w };
            unsigned short hh[4], ll[4];
#pragma unroll
            for (int q = 0; q < 4; ++q) {
                hh[q] = f32_to_bf16_rne(fa[q]);
                ll[q] = f32_to_bf16_rne(fa[q] - bf16_bits_to_f32(hh[q]));
            }
            *(unsigned*)&ah_s[r][c0 + j]     = (unsigned)hh[0] | ((unsigned)hh[1] << 16);
            *(unsigned*)&ah_s[r][c0 + j + 2] = (unsigned)hh[2] | ((unsigned)hh[3] << 16);
            *(unsigned*)&al_s[r][c0 + j]     = (unsigned)ll[0] | ((unsigned)ll[1] << 16);
            *(unsigned*)&al_s[r][c0 + j + 2] = (unsigned)ll[2] | ((unsigned)ll[3] << 16);
        }
    }
    __syncthreads();

    v8f c0v = {}; v8f c1v = {};
#pragma unroll
    for (int k0 = 0; k0 < 2 * DX; k0 += 32) {
        v16u ahi = load_fragA_bf16(&ah_s[0][0], 2 * DX, 0, k0, lane);
        v16u alo = load_fragA_bf16(&al_s[0][0], 2 * DX, 0, k0, lane);
        v16u b0h = load_fragB_bf16(w1h, NI, n0,      DX + k0, lane);
        v16u b0l = load_fragB_bf16(w1l, NI, n0,      DX + k0, lane);
        v16u b1h = load_fragB_bf16(w1h, NI, n0 + 16, DX + k0, lane);
        v16u b1l = load_fragB_bf16(w1l, NI, n0 + 16, DX + k0, lane);
        c0v = wmma_bf16(ahi, b0h, c0v);
        c0v = wmma_bf16(alo, b0h, c0v);
        c0v = wmma_bf16(ahi, b0l, c0v);
        c1v = wmma_bf16(ahi, b1h, c1v);
        c1v = wmma_bf16(alo, b1h, c1v);
        c1v = wmma_bf16(ahi, b1l, c1v);
    }

    int col = n0 + (lane & 15);
    int rb  = row0 + ((lane & 16) ? 8 : 0);
#pragma unroll
    for (int r = 0; r < 8; ++r) {
        p[(size_t)(rb + r) * NH + col]      = c0v[r];
        p[(size_t)(rb + r) * NH + col + 16] = c1v[r];
    }
}

// ---------------- Gather: agg[n] = sum over in-edges of p[src] ----------------
// One wave per node; lane covers 4 of 128 channels; register accumulation,
// zero float atomics. p (51 MB) is L2-resident -> pure L2-bandwidth phase.
__global__ void __launch_bounds__(256)
gather_kernel(const int* __restrict__ off, const int* __restrict__ csr,
              const float* __restrict__ p, float* __restrict__ agg, int n_nodes) {
    int gid  = blockIdx.x * 256 + threadIdx.x;
    int node = gid >> 5;
    int lane = gid & 31;
    if (node >= n_nodes) return;
    int b = off[node], e = off[node + 1];
    float4 acc = { 0.f, 0.f, 0.f, 0.f };
    for (int i = b; i < e; ++i) {
        int s = csr[i];
        float4 v = *(const float4*)(p + (size_t)s * NH + lane * 4);
        acc.x += v.x; acc.y += v.y; acc.z += v.z; acc.w += v.w;
    }
    *(float4*)(agg + (size_t)node * NH + lane * 4) = acc;
}

// ---------------- Phase C: hid = relu(x@W1x.T + agg + b1); out = hid@W2.T + b2 ----------------
__global__ void __launch_bounds__(128)
phaseC_kernel(const float* __restrict__ x, const float* __restrict__ agg,
              const unsigned short* __restrict__ w1h, const unsigned short* __restrict__ w1l,
              const unsigned short* __restrict__ w2h, const unsigned short* __restrict__ w2l,
              const float* __restrict__ b1, const float* __restrict__ b2,
              float* __restrict__ out) {
    __shared__ unsigned short sh_h[16][NH];   // x-tile, then hid-tile (hi)
    __shared__ unsigned short sh_l[16][NH];   // x-tile, then hid-tile (lo)

    int row0 = blockIdx.x * 16;
    int tid  = threadIdx.x;
    int wave = tid >> 5;
    int lane = tid & 31;
    int n0   = wave * 32;

    // stage & split the 16x128 x-tile into LDS (once per block)
    {
        int r  = tid >> 3;
        int c0 = (tid & 7) * 16;
        const float* sp = x + (size_t)(row0 + r) * DX + c0;
#pragma unroll
        for (int j = 0; j < 16; j += 4) {
            float4 f = *(const float4*)(sp + j);
            float fa[4] = { f.x, f.y, f.z, f.w };
            unsigned short hh[4], ll[4];
#pragma unroll
            for (int q = 0; q < 4; ++q) {
                hh[q] = f32_to_bf16_rne(fa[q]);
                ll[q] = f32_to_bf16_rne(fa[q] - bf16_bits_to_f32(hh[q]));
            }
            *(unsigned*)&sh_h[r][c0 + j]     = (unsigned)hh[0] | ((unsigned)hh[1] << 16);
            *(unsigned*)&sh_h[r][c0 + j + 2] = (unsigned)hh[2] | ((unsigned)hh[3] << 16);
            *(unsigned*)&sh_l[r][c0 + j]     = (unsigned)ll[0] | ((unsigned)ll[1] << 16);
            *(unsigned*)&sh_l[r][c0 + j + 2] = (unsigned)ll[2] | ((unsigned)ll[3] << 16);
        }
    }
    __syncthreads();

    // GEMM 1: x (hi/lo from LDS) x W1[:,0:128] (hi/lo), K = 128
    v8f c0v = {}; v8f c1v = {};
#pragma unroll
    for (int k0 = 0; k0 < DX; k0 += 32) {
        v16u ahi = load_fragA_bf16(&sh_h[0][0], NH, 0, k0, lane);
        v16u alo = load_fragA_bf16(&sh_l[0][0], NH, 0, k0, lane);
        v16u b0h = load_fragB_bf16(w1h, NI, n0,      k0, lane);
        v16u b0l = load_fragB_bf16(w1l, NI, n0,      k0, lane);
        v16u b1h = load_fragB_bf16(w1h, NI, n0 + 16, k0, lane);
        v16u b1l = load_fragB_bf16(w1l, NI, n0 + 16, k0, lane);
        c0v = wmma_bf16(ahi, b0h, c0v);
        c0v = wmma_bf16(alo, b0h, c0v);
        c0v = wmma_bf16(ahi, b0l, c0v);
        c1v = wmma_bf16(ahi, b1h, c1v);
        c1v = wmma_bf16(alo, b1h, c1v);
        c1v = wmma_bf16(ahi, b1l, c1v);
    }
    __syncthreads();   // everyone finished reading the x-tile

    // epilogue 1: + agg + b1, relu, split to bf16 hi/lo, overwrite LDS with hid
    int col = n0 + (lane & 15);
    int rb  = (lane & 16) ? 8 : 0;
    float bias0 = b1[col];
    float bias1 = b1[col + 16];
#pragma unroll
    for (int r = 0; r < 8; ++r) {
        int row = row0 + rb + r;
        float v0 = c0v[r] + agg[(size_t)row * NH + col]      + bias0;
        float v1 = c1v[r] + agg[(size_t)row * NH + col + 16] + bias1;
        v0 = v0 > 0.0f ? v0 : 0.0f;
        v1 = v1 > 0.0f ? v1 : 0.0f;
        unsigned short h0 = f32_to_bf16_rne(v0);
        unsigned short h1 = f32_to_bf16_rne(v1);
        sh_h[rb + r][col]      = h0;
        sh_h[rb + r][col + 16] = h1;
        sh_l[rb + r][col]      = f32_to_bf16_rne(v0 - bf16_bits_to_f32(h0));
        sh_l[rb + r][col + 16] = f32_to_bf16_rne(v1 - bf16_bits_to_f32(h1));
    }
    __syncthreads();

    // GEMM 2: hid (hi/lo from LDS) x W2 (hi/lo), K = 128
    v8f d0 = {}; v8f d1 = {};
#pragma unroll
    for (int k0 = 0; k0 < NH; k0 += 32) {
        v16u ah  = load_fragA_bf16(&sh_h[0][0], NH, 0, k0, lane);
        v16u al  = load_fragA_bf16(&sh_l[0][0], NH, 0, k0, lane);
        v16u b0h = load_fragB_bf16(w2h, NH, n0,      k0, lane);
        v16u b0l = load_fragB_bf16(w2l, NH, n0,      k0, lane);
        v16u b1h = load_fragB_bf16(w2h, NH, n0 + 16, k0, lane);
        v16u b1l = load_fragB_bf16(w2l, NH, n0 + 16, k0, lane);
        d0 = wmma_bf16(ah, b0h, d0);
        d0 = wmma_bf16(al, b0h, d0);
        d0 = wmma_bf16(ah, b0l, d0);
        d1 = wmma_bf16(ah, b1h, d1);
        d1 = wmma_bf16(al, b1h, d1);
        d1 = wmma_bf16(ah, b1l, d1);
    }

    float bb0 = b2[col];
    float bb1 = b2[col + 16];
#pragma unroll
    for (int r = 0; r < 8; ++r) {
        int row = row0 + rb + r;
        out[(size_t)row * NH + col]      = d0[r] + bb0;
        out[(size_t)row * NH + col + 16] = d1[r] + bb1;
    }
}

// ---------------- launch ----------------
extern "C" void kernel_launch(void* const* d_in, const int* in_sizes, int n_in,
                              void* d_out, int out_size, void* d_ws, size_t ws_size,
                              hipStream_t stream) {
    const float* x   = (const float*)d_in[0];
    const float* h   = (const float*)d_in[1];
    const int*   src = (const int*)d_in[2];
    const int*   dst = (const int*)d_in[3];
    const float* W1  = (const float*)d_in[4];
    const float* b1  = (const float*)d_in[5];
    const float* W2  = (const float*)d_in[6];
    const float* b2  = (const float*)d_in[7];
    float* out = (float*)d_out;

    const int n_nodes = in_sizes[0] / DX;     // 100000
    const int n_edges = in_sizes[2];          // 1600000

    // workspace layout
    float* p   = (float*)d_ws;                                  // n_nodes*128 f32
    float* agg = p + (size_t)n_nodes * NH;                      // n_nodes*128 f32
    unsigned short* w1h = (unsigned short*)(agg + (size_t)n_nodes * NH);
    unsigned short* w1l = w1h + NH * NI;
    unsigned short* w2h = w1l + NH * NI;
    unsigned short* w2l = w2h + NH * NH;
    int* cnt    = (int*)(w2l + NH * NH);                        // n_nodes
    int* off    = cnt + n_nodes;                                // n_nodes + 1
    int* cursor = off + (n_nodes + 1);                          // n_nodes
    int* csr    = cursor + n_nodes;                             // n_edges

    // 1) bf16 hi/lo weight split
    prep_w_kernel<<<(NH * NI + NH * NH + 255) / 256, 256, 0, stream>>>(
        W1, W2, w1h, w1l, w2h, w2l);

    // 2) CSR build: histogram -> exclusive scan -> slot fill
    zero_u32_kernel<<<(n_nodes + 255) / 256, 256, 0, stream>>>((unsigned*)cnt, n_nodes);
    hist_kernel<<<(n_edges + 255) / 256, 256, 0, stream>>>(dst, cnt, n_edges);
    scan_kernel<<<1, 1024, 0, stream>>>(cnt, off, cursor, n_nodes);
    fill_kernel<<<(n_edges + 255) / 256, 256, 0, stream>>>(src, dst, cursor, csr, n_edges);

    // 3) per-node message projection p = [x|h] @ W1m.T  (WMMA)
    int tiles = (n_nodes + 15) / 16;
    phaseA_kernel<<<tiles, 128, 0, stream>>>(x, h, w1h, w1l, p);

    // 4) atomic-free gather: agg[n] = sum_{e: dst=n} p[src[e]]
    size_t gthreads = (size_t)n_nodes * 32;
    gather_kernel<<<(unsigned)((gthreads + 255) / 256), 256, 0, stream>>>(
        off, csr, p, agg, n_nodes);

    // 5) fused MLP epilogue (WMMA)
    phaseC_kernel<<<tiles, 128, 0, stream>>>(x, agg, w1h, w1l, w2h, w2l, b1, b2, out);
}
